// InteractionBlock_22076131902170
// MI455X (gfx1250) — matrix operand order repaired and checked by reading, not compiled, not run on hardware
//
#include <hip/hip_runtime.h>

#define BB   2
#define NN   512
#define KB   16          // radial basis count
#define PTOT 592         // sum_l C_l*(2l+1) = 128+192+160+112
#define PPAD 640         // padded to 40 column tiles for 4-wide N blocking
#define CTOT 240         // sum_l C_l
#define JK   (NN*KB)     // 8192 contraction length
#define CK   512         // K-chunk staged in LDS for gemmB
#define NCHUNK (JK/CK)   // 16

typedef __bf16 bf16_t;
typedef __attribute__((ext_vector_type(16))) __bf16 v16bf;
typedef __attribute__((ext_vector_type(8)))  __bf16 v8bf;
typedef __attribute__((ext_vector_type(8)))  float  v8f;

struct W {
  const float* feats[4];
  const float* Wr1[4];
  const float* Wr2[4];
  const float* Wmix[4];
  const float* Ws[4];
  const float* bs[4];
};

// ---------------------------------------------------------------- K1: pairs
__global__ void pair_kernel(const float* __restrict__ pos,
                            const unsigned char* __restrict__ nmask,
                            bf16_t* __restrict__ RBF, bf16_t* __restrict__ Y) {
  size_t t = (size_t)blockIdx.x * blockDim.x + threadIdx.x;
  if (t >= (size_t)BB * NN * NN) return;
  int j = (int)(t % NN);
  int i = (int)((t / NN) % NN);
  int b = (int)(t / ((size_t)NN * NN));
  const float* pi = pos + ((size_t)b * NN + i) * 3;
  const float* pj = pos + ((size_t)b * NN + j) * 3;
  float rx = pj[0] - pi[0], ry = pj[1] - pi[1], rz = pj[2] - pi[2];
  float d = sqrtf(rx * rx + ry * ry + rz * rz + 1e-12f);
  float inv = 1.0f / d;
  float x = rx * inv, y = ry * inv, z = rz * inv;
  bool msk = (nmask[((size_t)b * NN + i) * NN + j] != 0) && (i != j);
  float env = (d < 5.0f) ? 0.5f * (__cosf(3.14159265358979f * d * 0.2f) + 1.0f) : 0.0f;
  float s = msk ? env : 0.0f;

  bf16_t* r = RBF + (((size_t)b * NN + i) * NN + j) * KB;
  const float gamma = 3.2f;                 // NUM_BASIS / R_C
  #pragma unroll
  for (int k = 0; k < KB; k++) {
    float mu = (5.0f / 15.0f) * (float)k;   // linspace(0, 5, 16)
    float dd = d - mu;
    r[k] = (bf16_t)(s * __expf(-gamma * dd * dd));
  }

  const float s3 = 1.7320508075688772f;
  float ysl[16];
  ysl[0] = 1.0f;
  ysl[1] = x; ysl[2] = y; ysl[3] = z;
  ysl[4] = s3 * x * y; ysl[5] = s3 * y * z; ysl[6] = 0.5f * (3.0f * z * z - 1.0f);
  ysl[7] = s3 * x * z; ysl[8] = 0.5f * s3 * (x * x - y * y);
  ysl[9]  = y * (3.0f * x * x - y * y);
  ysl[10] = x * y * z;
  ysl[11] = y * (5.0f * z * z - 1.0f);
  ysl[12] = z * (5.0f * z * z - 3.0f);
  ysl[13] = x * (5.0f * z * z - 1.0f);
  ysl[14] = z * (x * x - y * y);
  ysl[15] = x * (x * x - 3.0f * y * y);
  bf16_t* yo = Y + (((size_t)b * NN + i) * NN + j) * 16;
  #pragma unroll
  for (int q = 0; q < 16; q++) yo[q] = (bf16_t)ysl[q];
}

// ---------------------------------------------------------------- K2: h = f0 @ Wmix_l
__global__ void h_kernel(W w, float* __restrict__ hbuf) {
  int t = blockIdx.x * blockDim.x + threadIdx.x;
  if (t >= BB * NN * CTOT) return;
  int c = t % CTOT, bj = t / CTOT;
  int l, cl;
  if (c < 128)      { l = 0; cl = c; }
  else if (c < 192) { l = 1; cl = c - 128; }
  else if (c < 224) { l = 2; cl = c - 192; }
  else              { l = 3; cl = c - 224; }
  const int C[4] = {128, 64, 32, 16};
  const float* f0 = w.feats[0] + (size_t)bj * 128;   // feats0 [B,N,128,1]
  const float* wm = w.Wmix[l];
  float acc = 0.f;
  for (int c0 = 0; c0 < 128; c0++) acc += f0[c0] * wm[c0 * C[l] + cl];
  hbuf[(size_t)bj * CTOT + c] = acc;
}

// -------------------- K3: GT[b][p][jk] = Wr1[k,c]*feats[j,c,m]  (zero-padded cols)
__global__ void buildG_kernel(W w, bf16_t* __restrict__ GT) {
  size_t t = (size_t)blockIdx.x * blockDim.x + threadIdx.x;
  if (t >= (size_t)BB * PPAD * JK) return;
  int jk = (int)(t % JK);
  int p  = (int)((t / JK) % PPAD);
  int b  = (int)(t / ((size_t)JK * PPAD));
  if (p >= PTOT) { GT[t] = (bf16_t)0.0f; return; }
  int j = jk >> 4, k = jk & 15;
  int l, cl, m;
  if (p < 128)      { l = 0; cl = p; m = 0; }
  else if (p < 320) { int q = p - 128; l = 1; cl = q / 3; m = q % 3; }
  else if (p < 480) { int q = p - 320; l = 2; cl = q / 5; m = q % 5; }
  else              { int q = p - 480; l = 3; cl = q / 7; m = q % 7; }
  const int C[4] = {128, 64, 32, 16};
  int mc = 2 * l + 1;
  float wv = w.Wr1[l][k * C[l] + cl];
  float fv = w.feats[l][(((size_t)b * NN + j) * C[l] + cl) * mc + m];
  GT[t] = (bf16_t)(wv * fv);
}

// ------------------------------------------- K4: B2T[b][c][jk] = Wr2[k,c]*h[j,c]
__global__ void buildB2T_kernel(W w, const float* __restrict__ hbuf,
                                bf16_t* __restrict__ B2T) {
  size_t t = (size_t)blockIdx.x * blockDim.x + threadIdx.x;
  if (t >= (size_t)BB * CTOT * JK) return;
  int jk = (int)(t % JK);
  int c  = (int)((t / JK) % CTOT);
  int b  = (int)(t / ((size_t)JK * CTOT));
  int j = jk >> 4, k = jk & 15;
  int l, cl;
  if (c < 128)      { l = 0; cl = c; }
  else if (c < 192) { l = 1; cl = c - 128; }
  else if (c < 224) { l = 2; cl = c - 192; }
  else              { l = 3; cl = c - 224; }
  const int C[4] = {128, 64, 32, 16};
  float wv = w.Wr2[l][k * C[l] + cl];
  float hv = hbuf[((size_t)b * NN + j) * CTOT + c];
  B2T[t] = (bf16_t)(wv * hv);
}

// --------------- K5: path A WMMA GEMM: m[b] = RBF[b](512x8192) @ G[b](8192x592)
// Each wave: 16x64 output tile (4 accumulators), A fragment reused 4x.
// Register double-buffered: iteration k+32's fragments are in flight while
// iteration k's 4 WMMAs execute (avoids s_wait_loadcnt 0 full drains).
__global__ void gemmA_kernel(const bf16_t* __restrict__ RBF,
                             const bf16_t* __restrict__ GT,
                             float* __restrict__ mbuf) {
  int bid = blockIdx.x;
  int ng = bid % 10, mt = (bid / 10) % 32, b = bid / (10 * 32);
  int lane = threadIdx.x & 31;
  int lo16 = lane & 15;
  int grp  = (lane >> 4) & 1;
  // A: M = lane%16 (row i), 16 halves -> K base + grp*16 (contiguous)
  const bf16_t* arow = RBF + ((size_t)(b * NN) + mt * 16 + lo16) * JK + grp * 16;
  // B: N = lane%16 (col p), 16 halves -> K base + grp*16 (contiguous in GT row)
  const bf16_t* brow = GT + ((size_t)(b * PPAD) + ng * 64 + lo16) * JK + grp * 16;

  auto load_a = [&](int k0) -> v16bf {
    v8bf alo = *(const v8bf*)(arow + k0);
    v8bf ahi = *(const v8bf*)(arow + k0 + 8);
    v16bf a;
    #pragma unroll
    for (int h = 0; h < 8; h++) { a[h] = alo[h]; a[h + 8] = ahi[h]; }
    return a;
  };
  auto load_b = [&](int k0, int s) -> v16bf {
    return *(const v16bf*)(brow + (size_t)s * 16 * JK + k0);
  };

  v8f acc[4] = {{}, {}, {}, {}};
  v16bf a_cur = load_a(0);
  v16bf b_cur[4];
  #pragma unroll
  for (int s = 0; s < 4; s++) b_cur[s] = load_b(0, s);

  for (int k0 = 0; k0 < JK - 32; k0 += 32) {
    __builtin_prefetch(arow + k0 + 1024, 0, 3);   // WGP-scope prefetch of A stream
    v16bf a_nxt = load_a(k0 + 32);
    v16bf b_nxt[4];
    #pragma unroll
    for (int s = 0; s < 4; s++) b_nxt[s] = load_b(k0 + 32, s);
    #pragma unroll
    for (int s = 0; s < 4; s++)
      acc[s] = __builtin_amdgcn_wmma_f32_16x16x32_bf16(false, a_cur, false, b_cur[s],
                                                       (short)0, acc[s], false, false);
    a_cur = a_nxt;
    #pragma unroll
    for (int s = 0; s < 4; s++) b_cur[s] = b_nxt[s];
  }
  #pragma unroll
  for (int s = 0; s < 4; s++)
    acc[s] = __builtin_amdgcn_wmma_f32_16x16x32_bf16(false, a_cur, false, b_cur[s],
                                                     (short)0, acc[s], false, false);

  float* out = mbuf + (size_t)(b * NN) * PTOT;
  int rbase = mt * 16 + grp * 8;
  #pragma unroll
  for (int s = 0; s < 4; s++) {
    int col = ng * 64 + s * 16 + lo16;
    if (col < PTOT) {
      #pragma unroll
      for (int r = 0; r < 8; r++) out[(size_t)(rbase + r) * PTOT + col] = acc[s][r];
    }
  }
}

// --------------- K6: path B WMMA GEMM per (b,i): [240 x 8192] @ [8192 x 16]
// 4 waves/block share one B2T c-tile staged through LDS with async copies
// (double-buffered 16x512 chunks); B-operand (rbf*y) built on the fly per wave.
__global__ void __launch_bounds__(128)
gemmB_kernel(const bf16_t* __restrict__ RBF,
             const bf16_t* __restrict__ Y,
             const bf16_t* __restrict__ B2T,
             float* __restrict__ mbuf) {
  __shared__ bf16_t tileA[2][16][CK];   // 2 x 16KB double buffer

  int bid = blockIdx.x;
  int ct = bid % 15;
  int ig = (bid / 15) % (NN / 4);
  int b  = bid / (15 * (NN / 4));
  int tid  = threadIdx.x;
  int wid  = tid >> 5;
  int lane = tid & 31;
  int lo16 = lane & 15;
  int grp  = (lane >> 4) & 1;
  int i = ig * 4 + wid;

  const bf16_t* abase = B2T + ((size_t)(b * CTOT) + ct * 16) * JK;
  const bf16_t* rrow  = RBF + ((size_t)(b * NN) + i) * JK + grp * 16;
  const bf16_t* yrow  = Y + ((size_t)(b * NN) + i) * JK;   // [j][16] slots

  // stage one 16 x CK chunk: 1024 b128 async copies, 8 per thread
  auto issue_chunk = [&](int chunk, int buf) {
    int k0 = chunk * CK;
    #pragma unroll
    for (int q = 0; q < 8; q++) {
      int idx = q * 128 + tid;          // 0..1023
      int row = idx >> 6;               // 0..15
      int col = (idx & 63) * 8;         // halves, 16B granules
      const bf16_t* g = abase + (size_t)row * JK + k0 + col;
      unsigned lds = (unsigned)(uintptr_t)(&tileA[buf][row][col]);
      asm volatile("global_load_async_to_lds_b128 %0, %1, off"
                   :: "v"(lds), "v"(g) : "memory");
    }
  };

  v8f acc = {};
  issue_chunk(0, 0);
  for (int c = 0; c < NCHUNK; c++) {
    int buf = c & 1;
    if (c + 1 < NCHUNK) {
      issue_chunk(c + 1, buf ^ 1);
      asm volatile("s_wait_asynccnt 0x8" ::: "memory");  // chunk c landed
    } else {
      asm volatile("s_wait_asynccnt 0x0" ::: "memory");
    }
    __syncthreads();

    #pragma unroll 4
    for (int kk = 0; kk < CK; kk += 32) {
      v8bf alo = *(const v8bf*)&tileA[buf][lo16][kk + grp * 16];       // ds_load
      v8bf ahi = *(const v8bf*)&tileA[buf][lo16][kk + grp * 16 + 8];  // ds_load
      v16bf a;
      #pragma unroll
      for (int h = 0; h < 8; h++) { a[h] = alo[h]; a[h + 8] = ahi[h]; }
      int k0 = c * CK + kk;
      int j = (k0 + grp * 16) >> 4;                 // 16 K-halves share one j
      float yv = (float)yrow[j * 16 + lo16];        // N column = m slot
      v16bf rv = *(const v16bf*)(rrow + k0);
      v16bf bm;
      #pragma unroll
      for (int h = 0; h < 16; h++) bm[h] = (bf16_t)((float)rv[h] * yv);
      acc = __builtin_amdgcn_wmma_f32_16x16x32_bf16(false, a, false, bm,
                                                    (short)0, acc, false, false);
    }
    __syncthreads();
  }

  float* out = mbuf + ((size_t)(b * NN) + i) * PTOT;
  int col = lo16;
  int rbase = ct * 16 + grp * 8;
  const int Moff[4] = {0, 1, 4, 9};
  const int Poff[4] = {0, 128, 320, 480};
  #pragma unroll
  for (int r = 0; r < 8; r++) {
    int cc = rbase + r;
    int l, cl;
    if (cc < 128)      { l = 0; cl = cc; }
    else if (cc < 192) { l = 1; cl = cc - 128; }
    else if (cc < 224) { l = 2; cl = cc - 192; }
    else               { l = 3; cl = cc - 224; }
    int mc = 2 * l + 1;
    int ml = col - Moff[l];
    if (ml >= 0 && ml < mc) out[Poff[l] + cl * mc + ml] += acc[r];
  }
}

// ---------------------------------------------------------------- K7: dense + SiLU
__global__ void final_kernel(W w, const float* __restrict__ mbuf, float* __restrict__ out) {
  int t = blockIdx.x * blockDim.x + threadIdx.x;
  if (t >= BB * NN * PTOT) return;
  int q = t % PTOT, bi = t / PTOT;
  const int Qoff[4] = {0, 128, 320, 480};
  const int D[4] = {128, 192, 160, 112};
  int l;
  if (q < 128) l = 0; else if (q < 320) l = 1; else if (q < 480) l = 2; else l = 3;
  int ql = q - Qoff[l];
  const float* mrow = mbuf + (size_t)bi * PTOT + Qoff[l];
  const float* Ws = w.Ws[l];
  int d = D[l];
  float acc = 0.f;
  for (int p = 0; p < d; p++) acc += mrow[p] * Ws[p * d + ql];
  float xv = acc * (1.0f / NN) + w.bs[l][ql];      // m = (mA+mB)/N, then @Ws + bs
  out[t] = xv / (1.0f + __expf(-xv));              // silu
}

extern "C" void kernel_launch(void* const* d_in, const int* in_sizes, int n_in,
                              void* d_out, int out_size, void* d_ws, size_t ws_size,
                              hipStream_t stream) {
  (void)in_sizes; (void)n_in; (void)out_size; (void)ws_size;
  W w;
  for (int l = 0; l < 4; l++) w.feats[l] = (const float*)d_in[l];
  const float* pos = (const float*)d_in[4];
  const unsigned char* nmask = (const unsigned char*)d_in[5];
  for (int l = 0; l < 4; l++) {
    w.Wr1[l]  = (const float*)d_in[6 + 5 * l + 0];
    w.Wr2[l]  = (const float*)d_in[6 + 5 * l + 1];
    w.Wmix[l] = (const float*)d_in[6 + 5 * l + 2];
    w.Ws[l]   = (const float*)d_in[6 + 5 * l + 3];
    w.bs[l]   = (const float*)d_in[6 + 5 * l + 4];
  }
  char* ws = (char*)d_ws;
  size_t off = 0;
  auto take = [&](size_t bytes) -> char* {
    char* p = ws + off;
    off = (off + bytes + 255) & ~(size_t)255;
    return p;
  };
  bf16_t* RBF  = (bf16_t*)take((size_t)BB * NN * NN * KB * 2);   // ~16.8 MB
  bf16_t* Y    = (bf16_t*)take((size_t)BB * NN * NN * 16 * 2);   // ~16.8 MB
  bf16_t* GT   = (bf16_t*)take((size_t)BB * PPAD * JK * 2);      // ~21.0 MB
  bf16_t* B2T  = (bf16_t*)take((size_t)BB * CTOT * JK * 2);      // ~7.9 MB
  float*  hbuf = (float*)take((size_t)BB * NN * CTOT * 4);
  float*  mbuf = (float*)take((size_t)BB * NN * PTOT * 4);
  float*  outf = (float*)d_out;

  pair_kernel<<<(BB * NN * NN + 255) / 256, 256, 0, stream>>>(pos, nmask, RBF, Y);
  h_kernel<<<(BB * NN * CTOT + 255) / 256, 256, 0, stream>>>(w, hbuf);
  buildG_kernel<<<(int)(((size_t)BB * PPAD * JK + 255) / 256), 256, 0, stream>>>(w, GT);
  buildB2T_kernel<<<(int)(((size_t)BB * CTOT * JK + 255) / 256), 256, 0, stream>>>(w, hbuf, B2T);
  gemmA_kernel<<<BB * 32 * 10, 32, 0, stream>>>(RBF, GT, mbuf);
  gemmB_kernel<<<BB * 15 * (NN / 4), 128, 0, stream>>>(RBF, Y, B2T, mbuf);
  final_kernel<<<(BB * NN * PTOT + 255) / 256, 256, 0, stream>>>(w, mbuf, outf);
}
